// RSSM_40192303956009
// MI455X (gfx1250) — matrix operand order repaired and checked by reading, not compiled, not run on hardware
//
#include <hip/hip_runtime.h>

typedef __attribute__((ext_vector_type(16))) _Float16 v16h;
typedef __attribute__((ext_vector_type(8)))  float    v8f;

// ---------------------------------------------------------------------------
// Constants for this problem instance
// ---------------------------------------------------------------------------
#define BB   16          // batch (== WMMA M tile!)
#define TT   64
#define AA   18
#define DD   2048
#define HH   1024
#define EE   4096
#define SS   32
#define KK   32
#define SK   1024        // S*K
#define OUTW 6144        // 4*SK + D

// ---------------------------------------------------------------------------
// Pack fp32 row-major W[K][N] into WMMA-B fragment layout, f16, K zero-padded.
// Layout: Wp[((nt*kTiles + kt)*32 + lane)*16 + h]
//   n = nt*16 + (lane&15);  k = kt*32 + ((lane>=16)?16:0) + h
// ---------------------------------------------------------------------------
__global__ void rssm_pack_w(const float* __restrict__ W, int K, int N,
                            int kTiles, _Float16* __restrict__ Wp) {
  size_t idx = (size_t)blockIdx.x * blockDim.x + threadIdx.x;
  size_t total = (size_t)(N / 16) * kTiles * 512;
  if (idx >= total) return;
  int h    = (int)(idx & 15);
  int lane = (int)((idx >> 4) & 31);
  size_t t = idx >> 9;
  int kt = (int)(t % kTiles);
  int nt = (int)(t / kTiles);
  int k = kt * 32 + ((lane >> 4) << 4) + h;
  int n = nt * 16 + (lane & 15);
  Wp[idx] = (k < K) ? (_Float16)W[(size_t)k * N + n] : (_Float16)0.f;
}

// ---------------------------------------------------------------------------
// Pack the activation matrix A[16, K] = concat(X0[16,K0], X1[16,K1]) (f32)
// into WMMA-A fragment layout (f16, K zero-padded to kTiles*32). Runs once per
// GEMM; the A fragments are shared by every N tile.
// A layout: Ap[(kt*32 + lane)*16 + h],  m = lane&15, hi = lane>>4,
//           k = kt*32 + (hi<<4) + h
// ---------------------------------------------------------------------------
__global__ void rssm_pack_a(const float* __restrict__ X0, int ld0, int K0,
                            const float* __restrict__ X1, int ld1, int K1,
                            int kTiles, _Float16* __restrict__ Ap) {
  int idx = blockIdx.x * blockDim.x + threadIdx.x;
  if (idx >= kTiles * 512) return;
  int h    = idx & 15;
  int lane = (idx >> 4) & 31;
  int kt   = idx >> 9;
  int m  = lane & 15;
  int hi = lane >> 4;
  int k = kt * 32 + (hi << 4) + h;
  float v = 0.f;
  if (k < K0)           v = X0[m * ld0 + k];
  else if (k - K0 < K1) v = X1[m * ld1 + (k - K0)];
  Ap[idx] = (_Float16)v;
}

// ---------------------------------------------------------------------------
// WMMA GEMM: Y[16, N] = A @ W (+ bias), A and W pre-packed f16 fragments.
// One wave per 16-column N tile. Dual accumulators break the WMMA->WMMA RAW
// hazard chain (no v_nop slots); inner loop is pure b128 loads + 2 WMMAs.
// ---------------------------------------------------------------------------
__global__ __launch_bounds__(32)
void rssm_gemm16_wmma(const _Float16* __restrict__ Ap, int kTiles,
                      const _Float16* __restrict__ Wp,
                      const float* __restrict__ bias,
                      float* __restrict__ Y, int ldy) {
  const int nTile = blockIdx.x;
  const int lane  = threadIdx.x;        // 0..31
  const int m     = lane & 15;
  const int hi    = lane >> 4;

  v8f acc0 = {}, acc1 = {};
  const _Float16* ap = Ap + lane * 16;
  const _Float16* wp = Wp + (((size_t)nTile * kTiles) * 32 + lane) * 16;

  int kt = 0;
  for (; kt + 1 < kTiles; kt += 2) {
    v16h a0 = *(const v16h*)(ap);
    v16h b0 = *(const v16h*)(wp);
    v16h a1 = *(const v16h*)(ap + 512);
    v16h b1 = *(const v16h*)(wp + 512);
    ap += 1024;
    wp += 1024;
    __builtin_prefetch(wp, 0, 1);       // stream next weight tiles from L2
    acc0 = __builtin_amdgcn_wmma_f32_16x16x32_f16(
        false, a0, false, b0, (short)0, acc0, false, false);
    acc1 = __builtin_amdgcn_wmma_f32_16x16x32_f16(
        false, a1, false, b1, (short)0, acc1, false, false);
  }
  if (kt < kTiles) {                    // odd-kTiles tail (img_in: 33)
    v16h a = *(const v16h*)(ap);
    v16h b = *(const v16h*)(wp);
    acc0 = __builtin_amdgcn_wmma_f32_16x16x32_f16(
        false, a, false, b, (short)0, acc0, false, false);
  }

  // D layout: vgpr r -> row = r + hi*8, col = lane&15
  const int n  = nTile * 16 + m;
  const float bv = bias ? bias[n] : 0.f;
  float* yp = Y + (size_t)(hi << 3) * ldy + n;
#pragma unroll
  for (int r = 0; r < 8; ++r) {
    *yp = acc0[r] + acc1[r] + bv;
    yp += ldy;
  }
}

// ---------------------------------------------------------------------------
// LayerNorm (+optional SiLU) over last dim; one block per batch row; wave32
// shuffle tree + LDS cross-wave reduce. Safe in-place.
// ---------------------------------------------------------------------------
template <bool SILU>
__global__ __launch_bounds__(256)
void rssm_ln_act(const float* __restrict__ X, int N,
                 const float* __restrict__ g, const float* __restrict__ b,
                 float* __restrict__ Y) {
  const int row = blockIdx.x;
  const float* x = X + (size_t)row * N;
  float*       y = Y + (size_t)row * N;
  float s = 0.f, s2 = 0.f;
  for (int i = threadIdx.x; i < N; i += blockDim.x) {
    float v = x[i]; s += v; s2 += v * v;
  }
  __shared__ float sh[2][32];
  for (int o = 16; o; o >>= 1) { s += __shfl_xor(s, o); s2 += __shfl_xor(s2, o); }
  const int w = threadIdx.x >> 5, l = threadIdx.x & 31, nw = blockDim.x >> 5;
  if (l == 0) { sh[0][w] = s; sh[1][w] = s2; }
  __syncthreads();
  if (w == 0) {
    s  = (l < nw) ? sh[0][l] : 0.f;
    s2 = (l < nw) ? sh[1][l] : 0.f;
    for (int o = 16; o; o >>= 1) { s += __shfl_xor(s, o); s2 += __shfl_xor(s2, o); }
    if (l == 0) { sh[0][0] = s; sh[1][0] = s2; }
  }
  __syncthreads();
  const float mean = sh[0][0] / N;
  const float var  = sh[1][0] / N - mean * mean;
  const float inv  = rsqrtf(var + 1e-3f);
  for (int i = threadIdx.x; i < N; i += blockDim.x) {
    float v = (x[i] - mean) * inv * g[i] + b[i];
    if (SILU) v = v / (1.f + __expf(-v));
    y[i] = v;
  }
}

// ---------------------------------------------------------------------------
// GRU gate combine: parts = LN(concat(x,deter)@Wgru) in [16, 3*D].
// deter' = sigmoid(u-1)*tanh(sigmoid(r)*c) + (1-sigmoid(u-1))*deter
// Also streams deter' to the step output slot.
// ---------------------------------------------------------------------------
__global__ void rssm_gru_update(const float* __restrict__ parts,
                                float* __restrict__ deter,
                                float* __restrict__ out_deter, int outld) {
  int i = blockIdx.x * blockDim.x + threadIdx.x;
  if (i >= BB * DD) return;
  int b = i >> 11, j = i & (DD - 1);
  const float* p = parts + (size_t)b * (3 * DD);
  float r = 1.f / (1.f + __expf(-p[j]));
  float c = tanhf(r * p[DD + j]);
  float u = 1.f / (1.f + __expf(-(p[2 * DD + j] - 1.f)));
  float d = u * c + (1.f - u) * deter[i];
  deter[i] = d;
  out_deter[(size_t)b * outld + j] = d;
}

// ---------------------------------------------------------------------------
// Reset-on-is_first + action masking (start of every step).
// ---------------------------------------------------------------------------
__global__ void rssm_prep(const int* __restrict__ is_first, int t,
                          const float* __restrict__ action,
                          const float* __restrict__ deter0,
                          const float* __restrict__ stoch0,
                          float* __restrict__ deter,
                          float* __restrict__ stoch,
                          float* __restrict__ amask) {
  int i = blockIdx.x * blockDim.x + threadIdx.x;
  if (i >= BB * DD) return;
  int b = i >> 11, j = i & (DD - 1);
  float m = 1.f - (float)is_first[b * TT + t];
  deter[i] = deter[i] * m + deter0[i] * (1.f - m);
  if (j < SK) {
    int si = b * SK + j;
    stoch[si] = stoch[si] * m + stoch0[si] * (1.f - m);
  }
  if (j < AA) amask[b * AA + j] = action[((size_t)b * TT + t) * AA + j] * m;
}

// ---------------------------------------------------------------------------
// Fused softmax -> unimix -> log -> (Gumbel-)argmax one-hot. Forward value of
// the straight-through estimator is exactly the one-hot. One wave per (b,s),
// K=32 == wave32. Hash-based Gumbel (deterministic across graph replays).
// ---------------------------------------------------------------------------
__global__ __launch_bounds__(32)
void rssm_unimix_sample(const float* __restrict__ logits,
                        float* __restrict__ out_l,  // log-probs dest (or null)
                        float* __restrict__ out_oh, // one-hot dest  (or null)
                        int outld,
                        float* __restrict__ stoch,  // state dest (or null)
                        int use_gumbel, unsigned seed) {
  const int bs = blockIdx.x;            // b*32 + s
  const int b = bs >> 5, s = bs & 31;
  const int k = threadIdx.x;
  float x = logits[(size_t)b * SK + s * KK + k];
  float mx = x;
  for (int o = 16; o; o >>= 1) mx = fmaxf(mx, __shfl_xor(mx, o));
  float e = __expf(x - mx);
  float se = e;
  for (int o = 16; o; o >>= 1) se += __shfl_xor(se, o);
  float p = 0.99f * (e / se) + 0.01f / 32.f;
  float l = __logf(p);
  float score = l;
  if (use_gumbel) {
    unsigned h = seed ^ (unsigned)(bs * 1315423911u) ^ (unsigned)(k * 2654435761u);
    h ^= h >> 16; h *= 0x7feb352du; h ^= h >> 15; h *= 0x846ca68bu; h ^= h >> 16;
    float u = ((float)h + 1.0f) * 2.3283064e-10f;  // (0,1]
    score = l - __logf(-__logf(u));
  }
  float ms = score;
  for (int o = 16; o; o >>= 1) ms = fmaxf(ms, __shfl_xor(ms, o));
  unsigned long long ball = __ballot(score == ms);
  int win = __ffsll(ball) - 1;
  float oh = (k == win) ? 1.f : 0.f;
  size_t off = (size_t)b * outld + s * KK + k;
  if (out_l)  out_l[off]  = l;
  if (out_oh) out_oh[off] = oh;
  if (stoch)  stoch[(size_t)b * SK + s * KK + k] = oh;
}

// ---------------------------------------------------------------------------
// Learned initial deterministic state, broadcast over batch.
// ---------------------------------------------------------------------------
__global__ void rssm_init_deter(const float* __restrict__ Wi,
                                float* __restrict__ deter0) {
  int i = blockIdx.x * blockDim.x + threadIdx.x;
  if (i >= BB * DD) return;
  deter0[i] = tanhf(Wi[i & (DD - 1)]);
}

// ---------------------------------------------------------------------------
extern "C" void kernel_launch(void* const* d_in, const int* in_sizes, int n_in,
                              void* d_out, int out_size, void* d_ws, size_t ws_size,
                              hipStream_t stream) {
  const float* embed      = (const float*)d_in[0];
  const float* action     = (const float*)d_in[1];
  const int*   is_first   = (const int*)  d_in[2];
  const float* W_img_in   = (const float*)d_in[3];
  const float* g_img_in   = (const float*)d_in[4];
  const float* b_img_in   = (const float*)d_in[5];
  const float* W_gru      = (const float*)d_in[6];
  const float* g_gru      = (const float*)d_in[7];
  const float* b_gru      = (const float*)d_in[8];
  const float* W_img_out  = (const float*)d_in[9];
  const float* g_img_out  = (const float*)d_in[10];
  const float* b_img_out  = (const float*)d_in[11];
  const float* W_obs_out  = (const float*)d_in[12];
  const float* g_obs_out  = (const float*)d_in[13];
  const float* b_obs_out  = (const float*)d_in[14];
  const float* W_img_dist = (const float*)d_in[15];
  const float* bias_img   = (const float*)d_in[16];
  const float* W_obs_dist = (const float*)d_in[17];
  const float* bias_obs   = (const float*)d_in[18];
  const float* W_initial  = (const float*)d_in[19];
  float* out = (float*)d_out;
  (void)in_sizes; (void)n_in; (void)out_size; (void)ws_size;

  // k-tile counts (K padded to multiples of 32 in the packed layout)
  const int ktII = 33;   // 1042 -> 1056
  const int ktGR = 96;   // 3072
  const int ktIO = 64;   // 2048
  const int ktOO = 192;  // 6144
  const int ktDH = 32;   // 1024

  // ---- carve workspace ----
  char* ws = (char*)d_ws;
  size_t off = 0;
  auto carve = [&](size_t bytes) -> void* {
    off = (off + 255) & ~(size_t)255;
    void* p = ws + off;
    off += bytes;
    return p;
  };
  _Float16* Pii = (_Float16*)carve((size_t)64  * ktII * 512 * 2);
  _Float16* Pgr = (_Float16*)carve((size_t)384 * ktGR * 512 * 2);
  _Float16* Pio = (_Float16*)carve((size_t)64  * ktIO * 512 * 2);
  _Float16* Poo = (_Float16*)carve((size_t)64  * ktOO * 512 * 2);
  _Float16* Pid = (_Float16*)carve((size_t)64  * ktDH * 512 * 2);
  _Float16* Pod = (_Float16*)carve((size_t)64  * ktDH * 512 * 2);
  _Float16* Ap  = (_Float16*)carve((size_t)ktOO * 512 * 2);   // A fragments (max K)
  float* deter0 = (float*)carve((size_t)BB * DD * 4);
  float* stoch0 = (float*)carve((size_t)BB * SK * 4);
  float* deter  = (float*)carve((size_t)BB * DD * 4);
  float* stoch  = (float*)carve((size_t)BB * SK * 4);
  float* amask  = (float*)carve((size_t)BB * AA * 4);
  float* xbuf   = (float*)carve((size_t)BB * HH * 4);
  float* parts  = (float*)carve((size_t)BB * 3 * DD * 4);
  float* tmp    = (float*)carve((size_t)BB * HH * 4);
  float* logit  = (float*)carve((size_t)BB * SK * 4);

  // ---- pack all weights fp32 -> f16 WMMA-B fragments (one-time per call) ----
  auto pack = [&](const float* W, int K, int N, int kT, _Float16* P) {
    size_t total = (size_t)(N / 16) * kT * 512;
    rssm_pack_w<<<(unsigned)((total + 255) / 256), 256, 0, stream>>>(W, K, N, kT, P);
  };
  pack(W_img_in,   1042, HH,     ktII, Pii);
  pack(W_gru,      3072, 3 * DD, ktGR, Pgr);
  pack(W_img_out,  DD,   HH,     ktIO, Pio);
  pack(W_obs_out,  6144, HH,     ktOO, Poo);
  pack(W_img_dist, HH,   SK,     ktDH, Pid);
  pack(W_obs_dist, HH,   SK,     ktDH, Pod);

  // GEMM = pack A fragments once, then stream packed weights through WMMA.
  auto gemm = [&](const float* X0, int ld0, int K0,
                  const float* X1, int ld1, int K1,
                  int kT, const _Float16* P, const float* bias,
                  float* Y, int ldy, int nTiles) {
    rssm_pack_a<<<(kT * 512 + 255) / 256, 256, 0, stream>>>(
        X0, ld0, K0, X1, ld1, K1, kT, Ap);
    rssm_gemm16_wmma<<<nTiles, 32, 0, stream>>>(Ap, kT, P, bias, Y, ldy);
  };

  // ---- initial state: deter0 = tanh(W_initial); stoch0 = mode(prior(deter0))
  rssm_init_deter<<<(BB * DD + 255) / 256, 256, 0, stream>>>(W_initial, deter0);
  gemm(deter0, DD, DD, nullptr, 0, 0, ktIO, Pio, nullptr, tmp, HH, 64);
  rssm_ln_act<true><<<BB, 256, 0, stream>>>(tmp, HH, g_img_out, b_img_out, tmp);
  gemm(tmp, HH, HH, nullptr, 0, 0, ktDH, Pid, bias_img, logit, SK, 64);
  rssm_unimix_sample<<<BB * SS, 32, 0, stream>>>(logit, nullptr, nullptr, 0,
                                                 stoch0, /*gumbel=*/0, 0u);
  hipMemcpyAsync(deter, deter0, (size_t)BB * DD * 4, hipMemcpyDeviceToDevice, stream);
  hipMemcpyAsync(stoch, stoch0, (size_t)BB * SK * 4, hipMemcpyDeviceToDevice, stream);

  // ---- sequential scan over T ----
  for (int t = 0; t < TT; ++t) {
    float* outT = out + (size_t)t * BB * OUTW;

    // reset + action mask
    rssm_prep<<<(BB * DD + 255) / 256, 256, 0, stream>>>(
        is_first, t, action, deter0, stoch0, deter, stoch, amask);

    // x = SiLU(LN(concat(stoch, a) @ W_img_in))
    gemm(stoch, SK, SK, amask, AA, AA, ktII, Pii, nullptr, xbuf, HH, 64);
    rssm_ln_act<true><<<BB, 256, 0, stream>>>(xbuf, HH, g_img_in, b_img_in, xbuf);

    // GRU: parts = LN(concat(x, deter) @ W_gru); gate combine
    gemm(xbuf, HH, HH, deter, DD, DD, ktGR, Pgr, nullptr, parts, 3 * DD, 384);
    rssm_ln_act<false><<<BB, 256, 0, stream>>>(parts, 3 * DD, g_gru, b_gru, parts);
    rssm_gru_update<<<(BB * DD + 255) / 256, 256, 0, stream>>>(
        parts, deter, outT + 4096, OUTW);

    // prior head -> pl (out+2048), ps (out+3072)
    gemm(deter, DD, DD, nullptr, 0, 0, ktIO, Pio, nullptr, tmp, HH, 64);
    rssm_ln_act<true><<<BB, 256, 0, stream>>>(tmp, HH, g_img_out, b_img_out, tmp);
    gemm(tmp, HH, HH, nullptr, 0, 0, ktDH, Pid, bias_img, logit, SK, 64);
    rssm_unimix_sample<<<BB * SS, 32, 0, stream>>>(
        logit, outT + 2048, outT + 3072, OUTW, nullptr, /*gumbel=*/1,
        0x9E3779B9u * (unsigned)(2 * t) + 0x85EBCA6Bu);

    // posterior head -> ql (out+0), qs (out+1024), stoch
    gemm(deter, DD, DD, embed + (size_t)t * EE, TT * EE, EE,
         ktOO, Poo, nullptr, tmp, HH, 64);
    rssm_ln_act<true><<<BB, 256, 0, stream>>>(tmp, HH, g_obs_out, b_obs_out, tmp);
    gemm(tmp, HH, HH, nullptr, 0, 0, ktDH, Pod, bias_obs, logit, SK, 64);
    rssm_unimix_sample<<<BB * SS, 32, 0, stream>>>(
        logit, outT + 0, outT + 1024, OUTW, stoch, /*gumbel=*/1,
        0x9E3779B9u * (unsigned)(2 * t + 1) + 0x85EBCA6Bu);
  }
}
